// Spatial_MixFormer_30331059045128
// MI455X (gfx1250) — compile-verified, matrix operand order checked by reading, not run on hardware
//
#include <hip/hip_runtime.h>
#include <hip/hip_bf16.h>

typedef __attribute__((ext_vector_type(16))) _Float16 v16h;
typedef __attribute__((ext_vector_type(8)))  float    v8f;
typedef unsigned int v4u  __attribute__((ext_vector_type(4)));
typedef int          v8i_ __attribute__((ext_vector_type(8)));
typedef int          v4i_ __attribute__((ext_vector_type(4)));

#define N_      128
#define CIN     64
#define COUT    64
#define T_      256
#define V_      17
#define K_      3
#define GROUPS_ 8
#define O_      (K_*COUT)   // 192 conv outputs
#define VP      32          // padded v stride for m staging (halves)
#define POS     272         // 16 t * 17 v positions per tile

static const size_t M_ELEMS  = (size_t)N_*O_*T_*VP;     // f16 staging for m
static const size_t Y_ELEMS  = (size_t)N_*COUT*T_*V_;   // f32 y (pre-gate)
static const size_t QK_ELEMS = (size_t)N_*K_*4*2*V_;    // f32 QK pooling accum
static const size_t AT_ELEMS = (size_t)N_*K_*4*V_*V_;   // f32 atten chunks
// pooled: N_*COUT f32

// ---------------------------------------------------------------------------
// Kernel 1: 1x1 conv as WMMA GEMM  (conv_w[192x64] @ x0[n][64 x 272pos])
//   - conv_w staged via TDM TENSOR_LOAD_TO_LDS (TENSORcnt, one wave issues)
//   - x0 tile staged via GLOBAL_LOAD_ASYNC_TO_LDS_B32 (ASYNCcnt)
//   - fused QK channel/time pooling via LDS float atomics.
// grid = N_*16 blocks (one per (n, t-tile of 16)), 256 threads = 8 waves.
// ---------------------------------------------------------------------------
__global__ void __launch_bounds__(256) k_conv_wmma(
    const float* __restrict__ x0, const float* __restrict__ conv_w,
    const float* __restrict__ conv_b,
    _Float16* __restrict__ m_ws, float* __restrict__ qk_ws)
{
  const int tid  = threadIdx.x;
  const int wv   = tid >> 5;
  const int lane = tid & 31;
  const int n     = blockIdx.x >> 4;
  const int tbase = (blockIdx.x & 15) * 16;

  __shared__ alignas(32) _Float16 lds_w[O_*CIN];    // f16 A operand, 24 KB
  __shared__ alignas(32) float    lds_xf[POS*CIN];  // f32 async x0 staging, 68 KB
  // phase-shared buffer: f32 conv_w TDM staging (48 KB), then f16 x0 B operand
  __shared__ alignas(32) char     lds_u[O_*CIN*4];
  __shared__ float qk_s[K_*4*2*V_];                 // 408 floats
  __shared__ float cb_s[O_];

  float*    lds_wf = (float*)lds_u;      // phase 1 view
  _Float16* lds_x  = (_Float16*)lds_u;   // phase 2 view ([pos][c], 34 KB)

  // --- Issue async global->LDS copies for the x0 tile (f32, per-lane) ---
  for (int i = tid; i < POS*CIN; i += 256) {
    const int c = i / POS, pos = i % POS;
    const int t = tbase + pos / V_, v = pos % V_;
    const float* gp = &x0[(((size_t)n*CIN + c)*T_ + t)*V_ + v];
    const unsigned lds_off = (unsigned)(uintptr_t)&lds_xf[pos*CIN + c];
    const unsigned long long ga = (unsigned long long)(uintptr_t)gp;
    asm volatile("global_load_async_to_lds_b32 %0, %1, off"
                 :: "v"(lds_off), "v"(ga) : "memory");
  }

  // --- TDM: DMA the whole conv_w block (192x64 f32, contiguous) into LDS ---
#if __has_builtin(__builtin_amdgcn_tensor_load_to_lds)
  if (wv == 0) {
    const unsigned long long ga = (unsigned long long)(uintptr_t)conv_w;
    const unsigned lds_dst = (unsigned)(uintptr_t)lds_wf;
    const unsigned elems = O_*CIN;                  // 12288 f32 elements
    v4u g0 = 0;
    g0[0] = 1u;                                     // count=1, user descriptor
    g0[1] = lds_dst;                                // lds_addr
    g0[2] = (unsigned)(ga & 0xffffffffu);           // global_addr[31:0]
    g0[3] = (unsigned)((ga >> 32) & 0x01ffffffu)    // global_addr[56:32]
          | (2u << 30);                             // type = 2 (image)
    v8i_ g1 = 0;
    g1[0] = (int)(2u << 16);                        // data_size=2 (4 bytes)
    g1[1] = (int)((elems & 0xffffu) << 16);         // tensor_dim0[15:0]
    g1[2] = (int)(elems >> 16);                     // tensor_dim0[31:16]
    g1[3] = (int)((elems & 0xffffu) << 16);         // tile_dim0 (1D tile)
    g1[5] = (int)elems;                             // tensor_dim0_stride[31:0]
    v4i_ g2 = 0, g3 = 0;
#if __clang_major__ >= 23
    v8i_ g4 = 0;
    __builtin_amdgcn_tensor_load_to_lds(g0, g1, g2, g3, g4, 0);
#else
    __builtin_amdgcn_tensor_load_to_lds(g0, g1, g2, g3, 0);
#endif
    __builtin_amdgcn_s_wait_tensorcnt((short)0);
  }
#else
  for (int i = tid; i < O_*CIN; i += 256)
    lds_wf[i] = conv_w[i];
#endif

  // Overlap: small loads while the DMAs run.
  for (int i = tid; i < O_; i += 256)
    cb_s[i] = conv_b[i];
  for (int i = tid; i < K_*4*2*V_; i += 256)
    qk_s[i] = 0.f;

  asm volatile("s_wait_asynccnt 0x0" ::: "memory");
  __syncthreads();                          // lds_wf + lds_xf now valid

  // conv_w f32 -> f16 A-operand conversion
  for (int i = tid; i < O_*CIN; i += 256)
    lds_w[i] = (_Float16)lds_wf[i];
  __syncthreads();                          // all reads of lds_u (f32) done

  // x0 f32 -> f16 B-operand conversion (reuses lds_u)
  for (int i = tid; i < POS*CIN; i += 256)
    lds_x[i] = (_Float16)lds_xf[i];
  __syncthreads();

  const int ML = lane & 15;          // A/B row within tile
  const int KH = (lane >> 4) * 16;   // K-half selected by lane hi bit

  // 12 o-tiles x 17 pos-tiles, round-robin across 8 waves
  for (int tile = wv; tile < 12*17; tile += 8) {
    const int ot = tile / 17, pt = tile % 17;
    v8f acc = {};
    #pragma unroll
    for (int kb = 0; kb < CIN; kb += 32) {
      v16h a = *(const v16h*)&lds_w[(ot*16 + ML)*CIN + kb + KH];
      v16h b = *(const v16h*)&lds_x[(pt*16 + ML)*CIN + kb + KH];
      acc = __builtin_amdgcn_wmma_f32_16x16x32_f16(false, a, false, b,
                                                   (short)0, acc, false, false);
    }
    const int pos = pt*16 + ML;           // D: N = lane&15
    const int tl = pos / V_, v = pos % V_;
    #pragma unroll
    for (int j = 0; j < 8; ++j) {
      const int o = ot*16 + j + 8*(lane >> 4);   // D: M = j + 8*(lane>>4)
      const float val = acc[j] + cb_s[o];
      m_ws[(((size_t)n*O_ + o)*T_ + tbase + tl)*VP + v] = (_Float16)val;
      // QK pooling: channel c -> (j_attn = c>>4, sel = (c>>3)&1), mean over 8ch x T
      const int c = o & 63;
      const int qi = (((o >> 6)*4 + (c >> 4))*2 + ((c >> 3) & 1))*V_ + v;
      atomicAdd(&qk_s[qi], val);
    }
  }

  // zero the v=17..31 pad halves so kernel 3 can run K=32 WMMA cleanly
  for (int r = tid; r < O_*16; r += 256) {
    const int o = r >> 4, tl = r & 15;
    _Float16* p = &m_ws[(((size_t)n*O_ + o)*T_ + tbase + tl)*VP + V_];
    #pragma unroll
    for (int z = 0; z < VP - V_; ++z) p[z] = (_Float16)0.f;
  }
  __syncthreads();
  for (int i = tid; i < K_*4*2*V_; i += 256)
    atomicAdd(&qk_ws[(size_t)n*(K_*4*2*V_) + i], qk_s[i]);
}

// ---------------------------------------------------------------------------
// Kernel 2: per-n attention (Q/K linears, S & X softmaxes, chunk mixing).
// grid = N_ blocks, 256 threads. All tiny VALU work.
// ---------------------------------------------------------------------------
__global__ void __launch_bounds__(256) k_atten(
    const float* __restrict__ qk_ws,
    const float* __restrict__ spa_w, const float* __restrict__ spa_b,
    const float* __restrict__ mix_w, const float* __restrict__ mix_b,
    float* __restrict__ atten_ws)
{
  const int n = blockIdx.x, tid = threadIdx.x;
  __shared__ float Qo[12*V_], Ko[12*V_];
  __shared__ float QT[12*V_], KT[12*V_], Qc[12*V_], Kc[12*V_];
  __shared__ float Ss[12*V_*V_];   // S rows (kj, v, w)
  __shared__ float Xs[9*V_*V_];    // X rows (k, jx, v, w)

  for (int i = tid; i < 408; i += 256) {
    float val = qk_ws[(size_t)n*408 + i] * (1.0f/(8.0f*(float)T_));
    int vv = i % V_; int r = i / V_; int sel = r & 1; r >>= 1;   // r = kj
    if (sel == 0) Qo[r*V_+vv] = val; else Ko[r*V_+vv] = val;
  }
  __syncthreads();

  for (int i = tid; i < 12*V_; i += 256) {
    const int row = i / V_, v = i % V_;
    float s1 = spa_b[v], s2 = spa_b[v], m1 = mix_b[v], m2 = mix_b[v];
    for (int u = 0; u < V_; ++u) {
      const float ws = spa_w[v*V_+u], wm = mix_w[v*V_+u];
      s1 += Qo[row*V_+u]*ws; s2 += Ko[row*V_+u]*ws;
      m1 += Qo[row*V_+u]*wm; m2 += Ko[row*V_+u]*wm;
    }
    QT[i] = fmaxf(s1, 0.f); KT[i] = fmaxf(s2, 0.f);
    Qc[i] = fmaxf(m1, 0.f); Kc[i] = fmaxf(m2, 0.f);
  }
  __syncthreads();

  // S = softmax_w(QT[v]*KT[w]) per (kj, v)
  for (int i = tid; i < 12*V_; i += 256) {
    const int kj = i / V_, v = i % V_;
    const float q = QT[kj*V_+v];
    float row[V_], mx = -1e30f;
    for (int w = 0; w < V_; ++w) { row[w] = q*KT[kj*V_+w]; mx = fmaxf(mx, row[w]); }
    float sum = 0.f;
    for (int w = 0; w < V_; ++w) { row[w] = __expf(row[w]-mx); sum += row[w]; }
    const float inv = 1.f/sum;
    for (int w = 0; w < V_; ++w) Ss[i*V_+w] = row[w]*inv;
  }
  // X = softmax_w(Kc[k][jx][v]*Qc[k][jx+1][w]) for jx in 0..2
  for (int i = tid; i < 9*V_; i += 256) {
    const int r = i / V_, v = i % V_;
    const int k = r / 3, jx = r % 3;
    const float q = Kc[(k*4+jx)*V_+v];
    float row[V_], mx = -1e30f;
    for (int w = 0; w < V_; ++w) { row[w] = q*Qc[(k*4+jx+1)*V_+w]; mx = fmaxf(mx, row[w]); }
    float sum = 0.f;
    for (int w = 0; w < V_; ++w) { row[w] = __expf(row[w]-mx); sum += row[w]; }
    const float inv = 1.f/sum;
    for (int w = 0; w < V_; ++w) Xs[i*V_+w] = row[w]*inv;
  }
  __syncthreads();

  // atten chunks m1..m4
  for (int i = tid; i < K_*4*V_*V_; i += 256) {
    const int w = i % V_; int r = i / V_; const int v = r % V_; r /= V_;
    const int jj = r % 4, k = r / 4;
    float val = 0.5f * Ss[((k*4+jj)*V_+v)*V_+w];
    if (jj >= 1) val += 0.25f * Xs[((k*3 + (jj-1))*V_+v)*V_+w];
    if (jj <= 2) val += 0.25f * Xs[((k*3 + jj)*V_+v)*V_+w];
    atten_ws[(size_t)n*(K_*4*V_*V_) + i] = val;
  }
}

// ---------------------------------------------------------------------------
// Kernel 3: y[n,c] = sum_k m[n,k,c](256x17) @ A_final[n,k,c](17x17) via WMMA,
//           plus pooled[n,c] = mean(y) via LDS tree reduction.
// grid = N_*COUT blocks, 256 threads = 8 waves.
// ---------------------------------------------------------------------------
__global__ void __launch_bounds__(256) k_spatial_wmma(
    const _Float16* __restrict__ m_ws, const float* __restrict__ atten_ws,
    const float* __restrict__ A_GEME, const float* __restrict__ A_SE,
    float* __restrict__ y_ws, float* __restrict__ pooled_ws)
{
  const int tid = threadIdx.x, wv = tid >> 5, lane = tid & 31;
  const int n = blockIdx.x >> 6, c = blockIdx.x & 63;
  const int jj = c >> 4, g = c & 7;

  __shared__ alignas(32) _Float16 Af[K_*32*32];  // B operand, [k][w][v] padded
  __shared__ float red[256];

  for (int i = tid; i < K_*32*32; i += 256) Af[i] = (_Float16)0.f;
  __syncthreads();
  for (int i = tid; i < K_*V_*V_; i += 256) {
    const int k = i / (V_*V_); const int r = i % (V_*V_);
    const int v = r / V_, w = r % V_;
    const float at = atten_ws[(size_t)n*(K_*4*V_*V_) + ((k*4+jj)*V_ + v)*V_ + w];
    const size_t ai = ((size_t)(k*GROUPS_ + g)*V_ + v)*V_ + w;
    const float nm = A_SE[ai] + A_GEME[ai];
    Af[(k*32 + w)*32 + v] = (_Float16)(0.5f*at + nm);   // w-major: K contiguous
  }
  __syncthreads();

  const int ML = lane & 15, KH = (lane >> 4) * 16;
  float psum = 0.f;

  for (int tt = wv; tt < 16; tt += 8) {
    v8f acc0 = {}, acc1 = {};
    #pragma unroll
    for (int k = 0; k < K_; ++k) {
      const _Float16* ap =
          &m_ws[(((size_t)n*O_ + k*COUT + c)*T_ + tt*16 + ML)*VP + KH];
      if (k + 1 < K_)
        __builtin_prefetch(ap + (size_t)COUT*T_*VP, 0, 1);  // global_prefetch_b8
      v16h a  = *(const v16h*)ap;
      v16h b0 = *(const v16h*)&Af[(k*32 +      ML)*32 + KH];
      v16h b1 = *(const v16h*)&Af[(k*32 + 16 + ML)*32 + KH];
      acc0 = __builtin_amdgcn_wmma_f32_16x16x32_f16(false, a, false, b0,
                                                    (short)0, acc0, false, false);
      acc1 = __builtin_amdgcn_wmma_f32_16x16x32_f16(false, a, false, b1,
                                                    (short)0, acc1, false, false);
    }
    #pragma unroll
    for (int j = 0; j < 8; ++j) {
      const int trow = tt*16 + j + 8*(lane >> 4);
      const int w0 = lane & 15;
      const float v0 = acc0[j];
      y_ws[(((size_t)n*COUT + c)*T_ + trow)*V_ + w0] = v0;
      psum += v0;
      if (w0 == 0) {                      // acc1 column 0 == w=16 (rest is pad)
        const float v1 = acc1[j];
        y_ws[(((size_t)n*COUT + c)*T_ + trow)*V_ + 16] = v1;
        psum += v1;
      }
    }
  }

  red[tid] = psum;
  __syncthreads();
  for (int s = 128; s > 0; s >>= 1) {
    if (tid < s) red[tid] += red[tid + s];
    __syncthreads();
  }
  if (tid == 0)
    pooled_ws[(size_t)n*COUT + c] = red[0] * (1.0f/((float)T_*(float)V_));
}

// ---------------------------------------------------------------------------
// Kernel 4: channel gate + BN affine + residual + ReLU (elementwise).
// ---------------------------------------------------------------------------
__global__ void __launch_bounds__(256) k_final(
    const float* __restrict__ y_ws, const float* __restrict__ pooled_ws,
    const float* __restrict__ x0, const float* __restrict__ cha_w,
    const float* __restrict__ bn_gamma, const float* __restrict__ bn_beta,
    float* __restrict__ out)
{
  const size_t idx = (size_t)blockIdx.x*256 + threadIdx.x;
  const size_t total = (size_t)N_*COUT*T_*V_;
  if (idx >= total) return;
  const int c = (int)((idx / ((size_t)T_*V_)) % COUT);
  const int n = (int)(idx / ((size_t)COUT*T_*V_));
  const float* p = &pooled_ws[(size_t)n*COUT];
  const float pm1 = (c > 0)        ? p[c-1] : 0.f;
  const float pp1 = (c < COUT-1)   ? p[c+1] : 0.f;
  const float cr  = cha_w[0]*pm1 + cha_w[1]*p[c] + cha_w[2]*pp1;
  const float sig = 1.f / (1.f + __expf(-cr));
  const float y   = y_ws[idx];
  const float sc  = bn_gamma[c] * rsqrtf(1.0f + 1e-5f);
  const float r   = (y + y*sig) * sc + bn_beta[c] + x0[idx];
  out[idx] = fmaxf(r, 0.f);
}

// ---------------------------------------------------------------------------
extern "C" void kernel_launch(void* const* d_in, const int* in_sizes, int n_in,
                              void* d_out, int out_size, void* d_ws, size_t ws_size,
                              hipStream_t stream) {
  const float* x0       = (const float*)d_in[0];
  const float* conv_w   = (const float*)d_in[1];
  const float* conv_b   = (const float*)d_in[2];
  const float* spa_w    = (const float*)d_in[3];
  const float* spa_b    = (const float*)d_in[4];
  const float* mix_w    = (const float*)d_in[5];
  const float* mix_b    = (const float*)d_in[6];
  const float* A_GEME   = (const float*)d_in[7];
  const float* A_SE     = (const float*)d_in[8];
  const float* cha_w    = (const float*)d_in[9];
  const float* bn_gamma = (const float*)d_in[10];
  const float* bn_beta  = (const float*)d_in[11];

  char* ws = (char*)d_ws;
  _Float16* m_ws  = (_Float16*)ws;         size_t off  = M_ELEMS*sizeof(_Float16);
  float* y_ws     = (float*)(ws + off);    off += Y_ELEMS*sizeof(float);
  float* qk_ws    = (float*)(ws + off);    off += QK_ELEMS*sizeof(float);
  float* atten_ws = (float*)(ws + off);    off += AT_ELEMS*sizeof(float);
  float* pooled_ws= (float*)(ws + off);

  hipMemsetAsync(qk_ws, 0, QK_ELEMS*sizeof(float), stream);

  k_conv_wmma   <<<N_*16,   256, 0, stream>>>(x0, conv_w, conv_b, m_ws, qk_ws);
  k_atten       <<<N_,      256, 0, stream>>>(qk_ws, spa_w, spa_b, mix_w, mix_b, atten_ws);
  k_spatial_wmma<<<N_*COUT, 256, 0, stream>>>(m_ws, atten_ws, A_GEME, A_SE, y_ws, pooled_ws);

  const size_t total = (size_t)N_*COUT*T_*V_;
  k_final<<<(unsigned)((total + 255)/256), 256, 0, stream>>>(
      y_ws, pooled_ws, x0, cha_w, bn_gamma, bn_beta, (float*)d_out);
}
